// GAT_84052509983371
// MI455X (gfx1250) — matrix operand order repaired and checked by reading, not compiled, not run on hardware
//
#include <hip/hip_runtime.h>

// ---------------- CDNA5 WMMA types ----------------
typedef __attribute__((ext_vector_type(16))) __bf16 v16bf;
typedef __attribute__((ext_vector_type(8)))  float  v8f;

union FragBF { v16bf v; int4 q[2]; };

// ---------------- helpers ----------------
__device__ __forceinline__ unsigned short f2bf(float f) {
    unsigned int u = __float_as_uint(f);
    unsigned int r = ((u >> 16) & 1u) + 0x7FFFu;   // round-to-nearest-even
    return (unsigned short)((u + r) >> 16);
}

__device__ __forceinline__ float lrelu(float x) { return x > 0.f ? x : 0.2f * x; }

// float atomic max via monotone integer mapping (init with 0xFF800000 = -inf)
__device__ __forceinline__ void atomicMaxF(float* addr, float val) {
    if (val >= 0.f) atomicMax((int*)addr, __float_as_int(val));
    else            atomicMin((unsigned int*)addr, __float_as_uint(val));
}

__device__ __forceinline__ void edge_sd(const int* __restrict__ ei, int E, int e,
                                        int& s, int& d) {
    if (e < E) { s = ei[e]; d = ei[E + e]; }
    else       { s = e - E; d = s; }            // self-loop
}

// ---------------- utility kernels ----------------
__global__ void k_fill(unsigned int* __restrict__ p, unsigned int v, long n) {
    long i = (long)blockIdx.x * blockDim.x + threadIdx.x;
    if (i < n) p[i] = v;
}

__global__ void k_f2bf(const float* __restrict__ in, unsigned short* __restrict__ out, long n) {
    long i = (long)blockIdx.x * blockDim.x + threadIdx.x;
    if (i < n) out[i] = f2bf(in[i]);
}

// Swizzle W [K x ncols] (row-major, fp32) into per-wave B fragments:
// Ws[((tileN*nkb + kb)*32 + lane)*16 + j] = bf16(W[k, n]),
//   half = lane>>4, n = tileN*16 + (lane&15), k = kb*32 + half*16 + j
__global__ void k_prep_w(const float* __restrict__ W, unsigned short* __restrict__ Ws,
                         int nkb, int ncols, int total) {
    int tid = blockIdx.x * blockDim.x + threadIdx.x;
    if (tid >= total) return;
    int j = tid & 15;
    int t = tid >> 4;
    int lane = t & 31;
    t >>= 5;
    int kb    = t % nkb;
    int tileN = t / nkb;
    int k = kb * 32 + (lane >> 4) * 16 + j;
    int n = tileN * 16 + (lane & 15);
    Ws[tid] = f2bf(W[(size_t)k * ncols + n]);
}

// ---------------- WMMA GEMM: C[M x Ccols] = A[M x K](bf16) * B(swizzled bf16) ----------------
// one wave per 16x16 output tile; 8 waves per block
__global__ void k_gemm_bf16(const unsigned short* __restrict__ A,
                            const unsigned short* __restrict__ Bs,
                            float* __restrict__ C,
                            int M, int K, int Ccols, int ntN) {
    const int lane = threadIdx.x & 31;
    const int wave = threadIdx.x >> 5;
    const int tile = blockIdx.x * (blockDim.x >> 5) + wave;
    const int total = (M >> 4) * ntN;
    if (tile >= total) return;                  // uniform per wave: EXEC stays all-1s
    const int tileN = tile % ntN;
    const int tileM = tile / ntN;
    const int half  = lane >> 4;
    const int l15   = lane & 15;
    const int kbs   = K >> 5;

    const int4* arow  = (const int4*)(A + (size_t)(tileM * 16 + l15) * K);
    const int4* bbase = (const int4*)(Bs + (size_t)(tileN * kbs) * 32 * 16);

    v8f c = {};
    for (int kb = 0; kb < kbs; ++kb) {
        FragBF a, b;
        int o0 = (kb * 32 + half * 16) >> 3;    // int4 index within the A row
        a.q[0] = arow[o0];                      // K = kb*32 + half*16 + {0..7}
        a.q[1] = arow[o0 + 1];                  // K = kb*32 + half*16 + 8 + {0..7}
        const int4* bp = bbase + (size_t)(kb * 32 + lane) * 2;
        b.q[0] = bp[0];
        b.q[1] = bp[1];
        c = __builtin_amdgcn_wmma_f32_16x16x32_bf16(false, a.v, false, b.v,
                                                    (short)0, c, false, false);
    }
    float* crow = C + (size_t)(tileM * 16 + half * 8) * Ccols + tileN * 16 + l15;
#pragma unroll
    for (int r = 0; r < 8; ++r) crow[(size_t)r * Ccols] = c[r];
}

// ---------------- attention scores ----------------
// layer 1: thread per (node, head); h1 is [N, 8*32]
__global__ void k_scores1(const float* __restrict__ h1, const float* __restrict__ a_s,
                          const float* __restrict__ a_d, float* __restrict__ ss,
                          float* __restrict__ sd, int N) {
    int tid = blockIdx.x * blockDim.x + threadIdx.x;
    if (tid >= N * 8) return;
    int n = tid >> 3, hh = tid & 7;
    const float* hp = h1 + (size_t)n * 256 + hh * 32;
    const float* as = a_s + hh * 32;
    const float* ad = a_d + hh * 32;
    float s0 = 0.f, s1 = 0.f;
#pragma unroll
    for (int d = 0; d < 32; ++d) { float v = hp[d]; s0 += v * as[d]; s1 += v * ad[d]; }
    ss[tid] = s0; sd[tid] = s1;
}

// layer 2: thread per node; h2 is [N, 32], single head
__global__ void k_scores2(const float* __restrict__ h2, const float* __restrict__ a_s,
                          const float* __restrict__ a_d, float* __restrict__ ss,
                          float* __restrict__ sd, int N) {
    int n = blockIdx.x * blockDim.x + threadIdx.x;
    if (n >= N) return;
    const float* hp = h2 + (size_t)n * 32;
    float s0 = 0.f, s1 = 0.f;
#pragma unroll
    for (int d = 0; d < 32; ++d) { float v = hp[d]; s0 += v * a_s[d]; s1 += v * a_d[d]; }
    ss[n] = s0; sd[n] = s1;
}

// ---------------- edge softmax passes, layer 1 (H=8) ----------------
__global__ void k_edge_max1(const int* __restrict__ ei, int E, int ET,
                            const float* __restrict__ ss, const float* __restrict__ sd,
                            float* __restrict__ m) {
    int tid = blockIdx.x * blockDim.x + threadIdx.x;
    if (tid >= ET * 8) return;
    int e = tid >> 3, hh = tid & 7, s, d;
    edge_sd(ei, E, e, s, d);
    float ev = lrelu(ss[s * 8 + hh] + sd[d * 8 + hh]);
    atomicMaxF(&m[d * 8 + hh], ev);
}

__global__ void k_edge_den1(const int* __restrict__ ei, int E, int ET,
                            const float* __restrict__ ss, const float* __restrict__ sd,
                            const float* __restrict__ m, float* __restrict__ den) {
    int tid = blockIdx.x * blockDim.x + threadIdx.x;
    if (tid >= ET * 8) return;
    int e = tid >> 3, hh = tid & 7, s, d;
    edge_sd(ei, E, e, s, d);
    float ev = lrelu(ss[s * 8 + hh] + sd[d * 8 + hh]);
    atomicAdd(&den[d * 8 + hh], __expf(ev - m[d * 8 + hh]));
}

// one block (256 threads) per edge; thread c handles channel c, head = c>>5
__global__ void k_edge_acc1(const int* __restrict__ ei, int E,
                            const float* __restrict__ ss, const float* __restrict__ sd,
                            const float* __restrict__ m, const float* __restrict__ den,
                            const float* __restrict__ h1, float* __restrict__ acc) {
    int e = blockIdx.x;
    int c = threadIdx.x;
    int hh = c >> 5, s, d;
    edge_sd(ei, E, e, s, d);
    float ev = lrelu(ss[s * 8 + hh] + sd[d * 8 + hh]);
    float alpha = __expf(ev - m[d * 8 + hh]) / (den[d * 8 + hh] + 1e-16f);
    atomicAdd(&acc[(size_t)d * 256 + c], alpha * h1[(size_t)s * 256 + c]);
}

// out1 = relu(acc1 + b1) -> bf16 for the next GEMM
__global__ void k_bias_relu_bf16(const float* __restrict__ acc, const float* __restrict__ b,
                                 unsigned short* __restrict__ hb, long n) {
    long i = (long)blockIdx.x * blockDim.x + threadIdx.x;
    if (i >= n) return;
    float v = acc[i] + b[i & 255];
    hb[i] = f2bf(v > 0.f ? v : 0.f);
}

// ---------------- edge softmax passes, layer 2 (H=1) ----------------
__global__ void k_edge_max2(const int* __restrict__ ei, int E, int ET,
                            const float* __restrict__ ss, const float* __restrict__ sd,
                            float* __restrict__ m) {
    int e = blockIdx.x * blockDim.x + threadIdx.x;
    if (e >= ET) return;
    int s, d; edge_sd(ei, E, e, s, d);
    atomicMaxF(&m[d], lrelu(ss[s] + sd[d]));
}

__global__ void k_edge_den2(const int* __restrict__ ei, int E, int ET,
                            const float* __restrict__ ss, const float* __restrict__ sd,
                            const float* __restrict__ m, float* __restrict__ den) {
    int e = blockIdx.x * blockDim.x + threadIdx.x;
    if (e >= ET) return;
    int s, d; edge_sd(ei, E, e, s, d);
    atomicAdd(&den[d], __expf(lrelu(ss[s] + sd[d]) - m[d]));
}

// 256 threads = 8 edges x 32 channels
__global__ void k_edge_acc2(const int* __restrict__ ei, int E, int ET,
                            const float* __restrict__ ss, const float* __restrict__ sd,
                            const float* __restrict__ m, const float* __restrict__ den,
                            const float* __restrict__ h2, float* __restrict__ acc) {
    int e = blockIdx.x * 8 + (threadIdx.x >> 5);
    if (e >= ET) return;
    int c = threadIdx.x & 31, s, d;
    edge_sd(ei, E, e, s, d);
    float alpha = __expf(lrelu(ss[s] + sd[d]) - m[d]) / (den[d] + 1e-16f);
    atomicAdd(&acc[(size_t)d * 32 + c], alpha * h2[(size_t)s * 32 + c]);
}

__global__ void k_final(const float* __restrict__ acc, const float* __restrict__ b,
                        float* __restrict__ out, long n) {
    long i = (long)blockIdx.x * blockDim.x + threadIdx.x;
    if (i < n) out[i] = acc[i] + b[i & 31];
}

// ---------------- launcher ----------------
extern "C" void kernel_launch(void* const* d_in, const int* in_sizes, int n_in,
                              void* d_out, int out_size, void* d_ws, size_t ws_size,
                              hipStream_t stream) {
    const float* x   = (const float*)d_in[0];
    const int*   ei  = (const int*)  d_in[1];
    const float* W1  = (const float*)d_in[2];
    const float* a1s = (const float*)d_in[3];
    const float* a1d = (const float*)d_in[4];
    const float* b1  = (const float*)d_in[5];
    const float* W2  = (const float*)d_in[6];
    const float* a2s = (const float*)d_in[7];
    const float* a2d = (const float*)d_in[8];
    const float* b2  = (const float*)d_in[9];

    const int N  = in_sizes[0] / 128;   // 50000
    const int E  = in_sizes[1] / 2;     // 800000
    const int ET = E + N;               // + self loops

    // workspace carve-out (256B aligned regions)
    char* w = (char*)d_ws;
    auto carve = [&](size_t bytes) { char* p = w; w += (bytes + 255) & ~(size_t)255; return p; };
    unsigned short* xb   = (unsigned short*)carve((size_t)N * 128 * 2);
    unsigned short* W1s  = (unsigned short*)carve((size_t)16 * 4 * 32 * 16 * 2);
    unsigned short* W2s  = (unsigned short*)carve((size_t)2 * 8 * 32 * 16 * 2);
    float* h1   = (float*)carve((size_t)N * 256 * 4);
    float* s1s  = (float*)carve((size_t)N * 8 * 4);
    float* s1d  = (float*)carve((size_t)N * 8 * 4);
    float* m1   = (float*)carve((size_t)N * 8 * 4);
    float* den1 = (float*)carve((size_t)N * 8 * 4);
    float* acc1 = (float*)carve((size_t)N * 256 * 4);
    unsigned short* h1b = (unsigned short*)carve((size_t)N * 256 * 2);
    float* h2   = (float*)carve((size_t)N * 32 * 4);
    float* s2s  = (float*)carve((size_t)N * 4);
    float* s2d  = (float*)carve((size_t)N * 4);
    float* m2   = (float*)carve((size_t)N * 4);
    float* den2 = (float*)carve((size_t)N * 4);
    float* acc2 = (float*)carve((size_t)N * 32 * 4);

    const int TB = 256;
    auto blocks = [&](long n) { return (unsigned)((n + TB - 1) / TB); };

    // init accumulators (fresh every call -> deterministic)
    k_fill<<<blocks((long)N * 8),   TB, 0, stream>>>((unsigned*)m1,   0xFF800000u, (long)N * 8);
    k_fill<<<blocks((long)N * 8),   TB, 0, stream>>>((unsigned*)den1, 0u,          (long)N * 8);
    k_fill<<<blocks((long)N * 256), TB, 0, stream>>>((unsigned*)acc1, 0u,          (long)N * 256);
    k_fill<<<blocks(N),             TB, 0, stream>>>((unsigned*)m2,   0xFF800000u, N);
    k_fill<<<blocks(N),             TB, 0, stream>>>((unsigned*)den2, 0u,          N);
    k_fill<<<blocks((long)N * 32),  TB, 0, stream>>>((unsigned*)acc2, 0u,          (long)N * 32);

    // precision prep: x -> bf16, weights -> swizzled bf16 B-fragments
    k_f2bf<<<blocks((long)N * 128), TB, 0, stream>>>(x, xb, (long)N * 128);
    k_prep_w<<<blocks(16 * 4 * 32 * 16), TB, 0, stream>>>(W1, W1s, 4, 256, 16 * 4 * 32 * 16);
    k_prep_w<<<blocks(2 * 8 * 32 * 16),  TB, 0, stream>>>(W2, W2s, 8, 32,  2 * 8 * 32 * 16);

    // layer 1: h1 = x @ W1 (WMMA), scores, softmax-scatter
    {
        int tiles = (N / 16) * 16;               // 50000 tiles of 16x16
        k_gemm_bf16<<<(tiles + 7) / 8, 256, 0, stream>>>(xb, W1s, h1, N, 128, 256, 16);
    }
    k_scores1<<<blocks((long)N * 8), TB, 0, stream>>>(h1, a1s, a1d, s1s, s1d, N);
    k_edge_max1<<<blocks((long)ET * 8), TB, 0, stream>>>(ei, E, ET, s1s, s1d, m1);
    k_edge_den1<<<blocks((long)ET * 8), TB, 0, stream>>>(ei, E, ET, s1s, s1d, m1, den1);
    k_edge_acc1<<<ET, 256, 0, stream>>>(ei, E, s1s, s1d, m1, den1, h1, acc1);
    k_bias_relu_bf16<<<blocks((long)N * 256), TB, 0, stream>>>(acc1, b1, h1b, (long)N * 256);

    // layer 2: h2 = relu(out1) @ W2 (WMMA), scores, softmax-scatter
    {
        int tiles = (N / 16) * 2;                // 6250 tiles
        k_gemm_bf16<<<(tiles + 7) / 8, 256, 0, stream>>>(h1b, W2s, h2, N, 256, 32, 2);
    }
    k_scores2<<<blocks(N), TB, 0, stream>>>(h2, a2s, a2d, s2s, s2d, N);
    k_edge_max2<<<blocks(ET), TB, 0, stream>>>(ei, E, ET, s2s, s2d, m2);
    k_edge_den2<<<blocks(ET), TB, 0, stream>>>(ei, E, ET, s2s, s2d, m2, den2);
    k_edge_acc2<<<(ET + 7) / 8, 256, 0, stream>>>(ei, E, ET, s2s, s2d, m2, den2, h2, acc2);

    k_final<<<blocks((long)N * 32), TB, 0, stream>>>(acc2, b2, (float*)d_out, (long)N * 32);
}
